// ClusterProxyMemory_5033701671602
// MI455X (gfx1250) — matrix operand order repaired and checked by reading, not compiled, not run on hardware
//
#include <hip/hip_runtime.h>
#include <hip/hip_bf16.h>
#include <math.h>
#include <stdint.h>

typedef __attribute__((ext_vector_type(16))) __bf16 v16bf;
typedef __attribute__((ext_vector_type(8)))  __bf16 v8bf;
typedef __attribute__((ext_vector_type(4)))  __bf16 v4bf;
typedef __attribute__((ext_vector_type(8)))  float  v8f;
typedef __attribute__((ext_vector_type(4)))  float  v4f;
typedef __attribute__((ext_vector_type(4)))  unsigned int v4u;
typedef __attribute__((ext_vector_type(8)))  int    v8i;
typedef __attribute__((ext_vector_type(4)))  int    v4i;

#define NFEAT   2048
#define NSAMP   16384
#define NBATCH  256
#define NCHUNK  (NFEAT / 32)    // 64 K-chunks
#define INV_T   20.0f           // 1 / TEMP
#define SPECIAL 5554
#define IGNORE_IDX 1023         // SOURCE_CLASSES - 1
#define BSTR    40              // bf16 LDS row stride (80 B, 16B aligned)

// ---------------------------------------------------------------------------
// split one f32 vec4 into bf16 hi + bf16 lo (x ~= hi + lo, ~2^-16 rel. error)
// ---------------------------------------------------------------------------
__device__ __forceinline__ void split4(v4f x, __bf16* ph, __bf16* pl) {
  v4bf h, l;
  #pragma unroll
  for (int i = 0; i < 4; ++i) {
    __bf16 hh = (__bf16)x[i];
    h[i] = hh;
    l[i] = (__bf16)(x[i] - (float)hh);
  }
  *(v4bf*)ph = h;
  *(v4bf*)pl = l;
}

// ---------------------------------------------------------------------------
// TDM: DMA one 64-row x 32-col f32 tile (row stride 2048 f32) into LDS.
// D# built per CDNA5 ISA ch.8: group0 = {count, lds_addr, global_addr, type},
// group1 = {data_size, tensor/tile dims, dim0 stride}; 2-D -> groups 2/3 = 0.
// Issue from ONE wave only (EXEC ignored by tensor ops).
// This toolchain exposes the 6-arg builtin (extra int32x8 before cpol).
// ---------------------------------------------------------------------------
__device__ __forceinline__ void tdm_load_tile(const float* gsrc, void* lds_dst) {
  unsigned long long ga = (unsigned long long)(uintptr_t)gsrc;
  v4u g0;
  g0[0] = 1u;                                     // count=1, user mode
  g0[1] = (unsigned int)(uintptr_t)lds_dst;       // lds_addr (bytes)
  g0[2] = (unsigned int)ga;                       // global_addr[31:0]
  g0[3] = ((unsigned int)(ga >> 32) & 0x01FFFFFFu) | 0x80000000u; // [56:32]|type=2
  v8i g1;
  g1[0] = (int)(2u << 16);          // wg_mask=0 | data_size=2 (4 bytes)
  g1[1] = (int)(32u << 16);         // atomic_bar=0 | tensor_dim0=32 (lo16)
  g1[2] = (int)(64u << 16);         // tensor_dim0 hi=0 | tensor_dim1=64 (lo16)
  g1[3] = (int)(32u << 16);         // tensor_dim1 hi=0 | tile_dim0=32
  g1[4] = 64;                       // tile_dim1=64 | tile_dim2=0
  g1[5] = 2048;                     // tensor_dim0_stride[31:0] = 2048 elems
  g1[6] = 0;                        // stride0 hi | stride1 lo
  g1[7] = 0;                        // stride1 hi
  v4i z4 = {0, 0, 0, 0};
  v8i z8 = {0, 0, 0, 0, 0, 0, 0, 0};
  __builtin_amdgcn_tensor_load_to_lds(g0, g1, z4, z4, z8, 0);
}

// ---------------------------------------------------------------------------
// Pre-pass: split A [256x2048] f32 into global bf16 hi/lo (done once, tiny).
// ---------------------------------------------------------------------------
__global__ __launch_bounds__(256) void split_a_kernel(
    const float* __restrict__ A, __bf16* __restrict__ Ahi, __bf16* __restrict__ Alo)
{
  int i = blockIdx.x * 256 + threadIdx.x;        // float4 slot, 131072 total
  v4f x = ((const v4f*)A)[i];
  split4(x, Ahi + (size_t)i * 4, Alo + (size_t)i * 4);
}

// ---------------------------------------------------------------------------
// GEMM: C[m,n] = (1/T) * dot(A[m,:], B[n,:])
// Block = 16 waves (512 thr) covering ALL of M=256 x N=64; grid = 256 blocks.
// 3-stage pipeline: TDM(i+2) -> bf16-split(i+1) -> WMMA(i), double-buffered,
// one barrier per K-chunk. A fragments come pre-split from global bf16.
// ---------------------------------------------------------------------------
__global__ __launch_bounds__(512) void gemm_logits_kernel(
    const __bf16* __restrict__ Ahi,
    const __bf16* __restrict__ Alo,
    const float*  __restrict__ B,
    float* __restrict__ C)
{
  __shared__ __align__(16) float  Bf32[2][64 * 32];
  __shared__ __align__(16) __bf16 Bhi[2][64 * BSTR];
  __shared__ __align__(16) __bf16 Blo[2][64 * BSTR];

  const int tid  = threadIdx.x;
  const int lane = tid & 31;
  const int wm   = tid >> 5;            // wave id 0..15 -> M subtile
  const int half = (lane >> 4) & 1;     // half-wave selector
  const int lrow = lane & 15;
  const int n0   = blockIdx.x * 64;     // block N origin

  // conversion slot: 512 float4 slots (64 rows x 32 f32), one per thread
  const int srow = tid >> 3;
  const int sc4  = (tid & 7) << 2;

  const float* gBase = &B[(size_t)n0 * NFEAT];   // chunk kc tile = gBase + kc

  v8f acc[4] = {};

  // ---- prologue: DMA chunk 0, convert it, DMA chunk 1 ----
  if (tid < 32) {
    tdm_load_tile(gBase, &Bf32[0][0]);
    __builtin_amdgcn_s_wait_tensorcnt(0);
  }
  __syncthreads();
  {
    v4f x = *(const v4f*)&Bf32[0][srow * 32 + sc4];
    split4(x, &Bhi[0][srow * BSTR + sc4], &Blo[0][srow * BSTR + sc4]);
  }
  if (tid < 32) {
    tdm_load_tile(gBase + 32, &Bf32[1][0]);
    __builtin_amdgcn_s_wait_tensorcnt(0);
  }
  __syncthreads();

  int cb = 0, fb = 1;
  for (int kci = 0; kci < NCHUNK; ++kci) {
    const int kc = kci * 32;

    // 1. issue async DMA for chunk kci+2 (wave 0 only; EXEC-independent op)
    if (kci + 2 < NCHUNK && tid < 32)
      tdm_load_tile(gBase + kc + 64, &Bf32[fb ^ 1][0]);

    // 2. convert chunk kci+1 (f32 LDS -> bf16 hi/lo LDS)
    if (kci + 1 < NCHUNK) {
      v4f x = *(const v4f*)&Bf32[fb][srow * 32 + sc4];
      split4(x, &Bhi[cb ^ 1][srow * BSTR + sc4], &Blo[cb ^ 1][srow * BSTR + sc4]);
    }

    // 3. A fragments straight from global bf16 (16B loads, K-split layout)
    const size_t abase = (size_t)(wm * 16 + lrow) * NFEAT + kc + half * 8;
    v16bf a_hi, a_lo;
    *((v8bf*)&a_hi)     = *(const v8bf*)&Ahi[abase];
    *((v8bf*)&a_hi + 1) = *(const v8bf*)&Ahi[abase + 16];
    *((v8bf*)&a_lo)     = *(const v8bf*)&Alo[abase];
    *((v8bf*)&a_lo + 1) = *(const v8bf*)&Alo[abase + 16];

    // 4. 4 N-subtiles: bf16 fragments from LDS + 3 WMMAs each
    #pragma unroll
    for (int j = 0; j < 4; ++j) {
      const int boff = (j * 16 + lrow) * BSTR + half * 8;
      v16bf b_hi, b_lo;
      *((v8bf*)&b_hi)     = *(const v8bf*)&Bhi[cb][boff];
      *((v8bf*)&b_hi + 1) = *(const v8bf*)&Bhi[cb][boff + 16];
      *((v8bf*)&b_lo)     = *(const v8bf*)&Blo[cb][boff];
      *((v8bf*)&b_lo + 1) = *(const v8bf*)&Blo[cb][boff + 16];

      acc[j] = __builtin_amdgcn_wmma_f32_16x16x32_bf16(
                   false, a_hi, false, b_hi, (short)0, acc[j], false, false);
      acc[j] = __builtin_amdgcn_wmma_f32_16x16x32_bf16(
                   false, a_hi, false, b_lo, (short)0, acc[j], false, false);
      acc[j] = __builtin_amdgcn_wmma_f32_16x16x32_bf16(
                   false, a_lo, false, b_hi, (short)0, acc[j], false, false);
    }

    // 5. drain this iteration's TDM, then one barrier; swap buffers
    if (tid < 32)
      __builtin_amdgcn_s_wait_tensorcnt(0);
    __syncthreads();
    cb ^= 1;
    fb ^= 1;
  }

  // ---- store: C/D layout -> lanes 0-15: M=r, lanes 16-31: M=8+r; N=lrow ----
  const int mrow = wm * 16 + half * 8;
  #pragma unroll
  for (int j = 0; j < 4; ++j) {
    const int gcol = n0 + j * 16 + lrow;
    #pragma unroll
    for (int r = 0; r < 8; ++r) {
      C[(size_t)(mrow + r) * NSAMP + gcol] = acc[j][r] * INV_T;
    }
  }
}

// ---------------------------------------------------------------------------
// Row-wise logsumexp over 16384 logits (one block per row).
// ---------------------------------------------------------------------------
__global__ __launch_bounds__(256) void lse_kernel(
    const float* __restrict__ logits, float* __restrict__ lse)
{
  __shared__ float red[256];
  const int row = blockIdx.x;
  const int tid = threadIdx.x;
  const float* p = logits + (size_t)row * NSAMP;

  float m = -INFINITY;
  for (int i = tid; i < NSAMP; i += 256) m = fmaxf(m, p[i]);
  red[tid] = m; __syncthreads();
  for (int s = 128; s > 0; s >>= 1) {
    if (tid < s) red[tid] = fmaxf(red[tid], red[tid + s]);
    __syncthreads();
  }
  m = red[0]; __syncthreads();

  float sum = 0.0f;
  for (int i = tid; i < NSAMP; i += 256) sum += expf(p[i] - m);
  red[tid] = sum; __syncthreads();
  for (int s = 128; s > 0; s >>= 1) {
    if (tid < s) red[tid] += red[tid + s];
    __syncthreads();
  }
  if (tid == 0) lse[row] = m + logf(red[0]);
}

// ---------------------------------------------------------------------------
// Masked-mean NLL over the batch (single block, one thread per row).
// ---------------------------------------------------------------------------
__global__ __launch_bounds__(256) void loss_kernel(
    const int* __restrict__ targets,
    const float* __restrict__ logits,
    const float* __restrict__ lse,
    float* __restrict__ out)
{
  __shared__ float s_sum[256];
  __shared__ int   s_cnt[256];
  const int r = threadIdx.x;

  int t = targets[r] - 1;
  if (t == SPECIAL) t = IGNORE_IDX;
  const bool valid = (t >= 0) && (t != IGNORE_IDX);
  int tc = t < 0 ? 0 : (t > NSAMP - 1 ? NSAMP - 1 : t);
  const float nll = lse[r] - logits[(size_t)r * NSAMP + tc];

  s_sum[r] = valid ? nll : 0.0f;
  s_cnt[r] = valid ? 1 : 0;
  __syncthreads();
  for (int s = 128; s > 0; s >>= 1) {
    if (r < s) { s_sum[r] += s_sum[r + s]; s_cnt[r] += s_cnt[r + s]; }
    __syncthreads();
  }
  if (r == 0) {
    int n = s_cnt[0] > 1 ? s_cnt[0] : 1;
    out[0] = s_sum[0] / (float)n;
  }
}

// ---------------------------------------------------------------------------
extern "C" void kernel_launch(void* const* d_in, const int* in_sizes, int n_in,
                              void* d_out, int out_size, void* d_ws, size_t ws_size,
                              hipStream_t stream) {
  const float* inputs   = (const float*)d_in[0];   // [256 x 2048]
  const int*   targets  = (const int*)  d_in[1];   // [256]
  const float* features = (const float*)d_in[2];   // [16384 x 2048]
  float* out = (float*)d_out;

  // ws layout (16B-aligned slices):
  float*  logits = (float*)d_ws;                            // 256*16384 f32
  float*  lse    = logits + (size_t)NBATCH * NSAMP;         // 256 f32
  __bf16* Ahi    = (__bf16*)(lse + NBATCH);                 // 256*2048 bf16
  __bf16* Alo    = Ahi + (size_t)NBATCH * NFEAT;            // 256*2048 bf16

  split_a_kernel<<<(NBATCH * NFEAT / 4) / 256, 256, 0, stream>>>(inputs, Ahi, Alo);
  gemm_logits_kernel<<<NSAMP / 64, 512, 0, stream>>>(Ahi, Alo, features, logits);
  lse_kernel<<<NBATCH, 256, 0, stream>>>(logits, lse);
  loss_kernel<<<1, 256, 0, stream>>>(targets, logits, lse, out);
}
